// EdgeGCN_K_Sum_5076651344425
// MI455X (gfx1250) — compile-verified, matrix operand order checked by reading, not run on hardware
//
#include <hip/hip_runtime.h>
#include <hip/hip_fp16.h>

#define N_NODES   65536
#define N_EDGES   1048576
#define NODE_F    32
#define EDGE_F    8
#define HID       64
#define BATCH_SZ  64

typedef __attribute__((ext_vector_type(16))) _Float16 v16h;
typedef __attribute__((ext_vector_type(8)))  float    v8f;

__device__ __forceinline__ v8f wmma_16x16x32(v16h a, v16h b, v8f c) {
  // D = A(16x32 f16) x B(32x16 f16) + C(16x16 f32)
  return __builtin_amdgcn_wmma_f32_16x16x32_f16(false, a, false, b, (short)0, c,
                                                false, false);
}

// ---- fragment loaders (wave32, 16-bit A layout: lane = row M, K split by lane half) ----

// A fragment 16x32 (MxK) from row-major f16 source (global or LDS), lda elems/row.
__device__ __forceinline__ v16h load_a_f16(const _Float16* src, int lda, int lane) {
  int m  = lane & 15;
  int kb = (lane & 16) >> 1;              // 0 or 8
  const _Float16* p = src + m * lda + kb;
  v16h a;
#pragma unroll
  for (int j = 0; j < 8; ++j) a[j] = p[j];          // K = kb + 0..7
#pragma unroll
  for (int j = 0; j < 8; ++j) a[j + 8] = p[16 + j]; // K = 16 + kb + 0..7
  return a;
}

// A fragment from row-major f32 source, converting to f16; K >= kmax zero-padded.
__device__ __forceinline__ v16h load_a_f32(const float* src, int lda, int kmax, int lane) {
  int m  = lane & 15;
  int kb = (lane & 16) >> 1;
  v16h a;
#pragma unroll
  for (int j = 0; j < 8; ++j) {
    int k = kb + j;
    a[j] = (k < kmax) ? (_Float16)src[m * lda + k] : (_Float16)0.0f;
  }
#pragma unroll
  for (int j = 0; j < 8; ++j) {
    int k = 16 + kb + j;
    a[j + 8] = (k < kmax) ? (_Float16)src[m * lda + k] : (_Float16)0.0f;
  }
  return a;
}

// B fragment 32x16 (KxN): lane holds column n of B; tile at (k0, n0) of row-major
// f32 weight matrix W with ldn columns; rows >= kmax zero-padded.
__device__ __forceinline__ v16h load_b_f32(const float* W, int ldn, int k0, int n0,
                                           int kmax, int lane) {
  int n  = n0 + (lane & 15);
  int kb = (lane & 16) >> 1;
  v16h b;
#pragma unroll
  for (int j = 0; j < 8; ++j) {
    int k = k0 + kb + j;
    b[j] = (k < kmax) ? (_Float16)W[k * ldn + n] : (_Float16)0.0f;
  }
#pragma unroll
  for (int j = 0; j < 8; ++j) {
    int k = k0 + 16 + kb + j;
    b[j + 8] = (k < kmax) ? (_Float16)W[k * ldn + n] : (_Float16)0.0f;
  }
  return b;
}

// ---- kernel 1: support = node_features[65536,32] @ W_in[32,64] -> f16 ----
__global__ void node_gemm_in(const float* __restrict__ nodef,
                             const float* __restrict__ W_in,
                             _Float16* __restrict__ support) {
  int wave  = (int)((blockIdx.x * blockDim.x + threadIdx.x) >> 5);
  int lane  = threadIdx.x & 31;
  int node0 = wave * 16;
  if (node0 >= N_NODES) return;
  v16h a  = load_a_f32(nodef + (size_t)node0 * NODE_F, NODE_F, NODE_F, lane);
  int nl  = lane & 15;
  int mb  = (lane & 16) >> 1;
#pragma unroll
  for (int nt = 0; nt < 4; ++nt) {
    v16h b = load_b_f32(W_in, HID, 0, nt * 16, NODE_F, lane);
    v8f c = {};
    c = wmma_16x16x32(a, b, c);
#pragma unroll
    for (int r = 0; r < 8; ++r)
      support[(size_t)(node0 + mb + r) * HID + nt * 16 + nl] = (_Float16)c[r];
  }
}

// ---- kernel 2: fused edge layer (layers 1 & 2) ----
// Per 16-edge tile (one wave): enc = relu(ef@W1+b1)@W2+b2 computed via WMMA
// with h staged through LDS; msg = support_f16[Esrc]*enc scattered with f32 atomics.
__global__ void edge_layer(const float* __restrict__ ef,
                           const int* __restrict__ Esrc,
                           const int* __restrict__ Etgt,
                           const float* __restrict__ W1, const float* __restrict__ b1,
                           const float* __restrict__ W2, const float* __restrict__ b2,
                           const _Float16* __restrict__ support,
                           float* __restrict__ agg) {
  __shared__ _Float16 lds_h[8 * 16 * HID];   // 2KB per wave, wave-private
  int wid  = threadIdx.x >> 5;
  int lane = threadIdx.x & 31;
  int wave = blockIdx.x * 8 + wid;
  int e0   = wave * 16;
  _Float16* h = lds_h + wid * 16 * HID;

  int nl = lane & 15;
  int mb = (lane & 16) >> 1;

  // MLP layer 1: h = relu(ef_tile[16x8] @ W1[8x64] + b1), K padded 8->32
  v16h a1 = load_a_f32(ef + (size_t)e0 * EDGE_F, EDGE_F, EDGE_F, lane);
#pragma unroll
  for (int nt = 0; nt < 4; ++nt) {
    v16h b = load_b_f32(W1, HID, 0, nt * 16, EDGE_F, lane);
    v8f c = {};
    c = wmma_16x16x32(a1, b, c);
    float bias = b1[nt * 16 + nl];
#pragma unroll
    for (int r = 0; r < 8; ++r) {
      float v = c[r] + bias;
      h[(mb + r) * HID + nt * 16 + nl] = (_Float16)(v > 0.0f ? v : 0.0f);
    }
  }
  // wave-private region; enforce LDS store->load ordering
  asm volatile("s_wait_dscnt 0" ::: "memory");

  v16h a2lo = load_a_f16(h, HID, lane);       // K = 0..31
  v16h a2hi = load_a_f16(h + 32, HID, lane);  // K = 32..63

  int src[8], tgt[8];
#pragma unroll
  for (int r = 0; r < 8; ++r) {
    src[r] = Esrc[e0 + mb + r];
    tgt[r] = Etgt[e0 + mb + r];
  }

  // MLP layer 2 + gather-gate-scatter, one 16-wide N tile at a time
#pragma unroll
  for (int nt = 0; nt < 4; ++nt) {
    v16h blo = load_b_f32(W2, HID, 0,  nt * 16, HID, lane);
    v16h bhi = load_b_f32(W2, HID, 32, nt * 16, HID, lane);
    v8f c = {};
    c = wmma_16x16x32(a2lo, blo, c);
    c = wmma_16x16x32(a2hi, bhi, c);
    float bias = b2[nt * 16 + nl];
    int col = nt * 16 + nl;
#pragma unroll
    for (int r = 0; r < 8; ++r) {
      float enc = c[r] + bias;
      float msg = (float)support[(size_t)src[r] * HID + col] * enc;
      atomicAdd(&agg[(size_t)tgt[r] * HID + col], msg);
    }
  }
}

// ---- kernel 3: x = relu(agg + bias); support_out = x @ W[64x64] -> f16 ----
__global__ void node_update_gemm(const float* __restrict__ agg,
                                 const float* __restrict__ bias,
                                 const float* __restrict__ W,
                                 _Float16* __restrict__ support_out) {
  __shared__ _Float16 lds_x[8 * 16 * HID];
  int wid   = threadIdx.x >> 5;
  int lane  = threadIdx.x & 31;
  int wave  = blockIdx.x * 8 + wid;
  int node0 = wave * 16;
  _Float16* x = lds_x + wid * 16 * HID;

#pragma unroll
  for (int i = 0; i < 32; ++i) {
    int idx = lane + i * 32;              // 0..1023 over the 16x64 tile
    int k   = idx & (HID - 1);
    float v = agg[(size_t)node0 * HID + idx] + bias[k];
    x[idx]  = (_Float16)(v > 0.0f ? v : 0.0f);
  }
  asm volatile("s_wait_dscnt 0" ::: "memory");

  v16h alo = load_a_f16(x, HID, lane);
  v16h ahi = load_a_f16(x + 32, HID, lane);
  int nl = lane & 15;
  int mb = (lane & 16) >> 1;
#pragma unroll
  for (int nt = 0; nt < 4; ++nt) {
    v16h blo = load_b_f32(W, HID, 0,  nt * 16, HID, lane);
    v16h bhi = load_b_f32(W, HID, 32, nt * 16, HID, lane);
    v8f c = {};
    c = wmma_16x16x32(alo, blo, c);
    c = wmma_16x16x32(ahi, bhi, c);
#pragma unroll
    for (int r = 0; r < 8; ++r)
      support_out[(size_t)(node0 + mb + r) * HID + nt * 16 + nl] = (_Float16)c[r];
  }
}

// ---- kernel 4: layer-3 node projection (OUT_F = 1, scalar VALU) ----
__global__ void node_out_proj(const float* __restrict__ agg,
                              const float* __restrict__ b_mid,
                              const float* __restrict__ W_out,
                              float* __restrict__ sup_out) {
  int n = blockIdx.x * blockDim.x + threadIdx.x;
  if (n >= N_NODES) return;
  const float* row = agg + (size_t)n * HID;
  float acc = 0.0f;
#pragma unroll
  for (int k = 0; k < HID; ++k) {
    float v = row[k] + b_mid[k];
    v = v > 0.0f ? v : 0.0f;
    acc += v * W_out[k];
  }
  sup_out[n] = acc;
}

// ---- kernel 5: layer-3 edge gate + scatter (scalar) ----
__global__ void edge_out_layer(const float* __restrict__ ef,
                               const int* __restrict__ Esrc,
                               const int* __restrict__ Etgt,
                               const float* __restrict__ W1,  // [8]
                               const float* __restrict__ b1,  // [1]
                               const float* __restrict__ W2,  // [1]
                               const float* __restrict__ b2,  // [1]
                               const float* __restrict__ sup, // [N_NODES]
                               float* __restrict__ agg3) {
  int e = blockIdx.x * blockDim.x + threadIdx.x;
  if (e >= N_EDGES) return;
  const float* f = ef + (size_t)e * EDGE_F;
  float t = b1[0];
#pragma unroll
  for (int k = 0; k < EDGE_F; ++k) t += f[k] * W1[k];
  t = t > 0.0f ? t : 0.0f;
  float enc = t * W2[0] + b2[0];
  float msg = sup[Esrc[e]] * enc;
  atomicAdd(&agg3[Etgt[e]], msg);
}

// ---- kernel 6: batch sum-pool with LDS bins ----
__global__ void pool_kernel(const float* __restrict__ agg3,
                            const int* __restrict__ batch,
                            const float* __restrict__ b_out,
                            float* __restrict__ out) {
  __shared__ float bins[BATCH_SZ];
  int t = threadIdx.x;
  if (t < BATCH_SZ) bins[t] = 0.0f;
  __syncthreads();
  int n = blockIdx.x * blockDim.x + t;
  float v = agg3[n] + b_out[0];
  atomicAdd(&bins[batch[n]], v);
  __syncthreads();
  if (t < BATCH_SZ) atomicAdd(&out[t], bins[t]);
}

extern "C" void kernel_launch(void* const* d_in, const int* in_sizes, int n_in,
                              void* d_out, int out_size, void* d_ws, size_t ws_size,
                              hipStream_t stream) {
  const float* nodef     = (const float*)d_in[0];
  const float* edgef     = (const float*)d_in[1];
  const int*   Esrc      = (const int*)d_in[2];
  const int*   Etgt      = (const int*)d_in[3];
  const int*   batch     = (const int*)d_in[4];
  const float* W_in      = (const float*)d_in[5];
  const float* b_in      = (const float*)d_in[6];
  const float* W_mid     = (const float*)d_in[7];
  const float* b_mid     = (const float*)d_in[8];
  const float* W_out     = (const float*)d_in[9];
  const float* b_out     = (const float*)d_in[10];
  const float* ee_in_W1  = (const float*)d_in[11];
  const float* ee_in_b1  = (const float*)d_in[12];
  const float* ee_in_W2  = (const float*)d_in[13];
  const float* ee_in_b2  = (const float*)d_in[14];
  const float* ee_mid_W1 = (const float*)d_in[15];
  const float* ee_mid_b1 = (const float*)d_in[16];
  const float* ee_mid_W2 = (const float*)d_in[17];
  const float* ee_mid_b2 = (const float*)d_in[18];
  const float* ee_out_W1 = (const float*)d_in[19];
  const float* ee_out_b1 = (const float*)d_in[20];
  const float* ee_out_W2 = (const float*)d_in[21];
  const float* ee_out_b2 = (const float*)d_in[22];
  float* out = (float*)d_out;

  char* ws = (char*)d_ws;
  _Float16* support   = (_Float16*)(ws);                               // 8 MiB
  _Float16* support_m = (_Float16*)(ws + ((size_t)8  << 20));          // 8 MiB
  float*    agg       = (float*)   (ws + ((size_t)16 << 20));          // 16 MiB (reused)
  float*    sup_out   = (float*)   (ws + ((size_t)32 << 20));          // 256 KiB
  float*    agg3      = (float*)   (ws + ((size_t)32 << 20) + ((size_t)256 << 10));

  const size_t aggBytes = (size_t)N_NODES * HID * sizeof(float);

  // ---- layer 1 ----
  hipLaunchKernelGGL(node_gemm_in, dim3(512), dim3(256), 0, stream,
                     nodef, W_in, support);
  hipMemsetAsync(agg, 0, aggBytes, stream);
  hipLaunchKernelGGL(edge_layer, dim3(8192), dim3(256), 0, stream,
                     edgef, Esrc, Etgt, ee_in_W1, ee_in_b1, ee_in_W2, ee_in_b2,
                     support, agg);
  // ---- layer 2 ----
  hipLaunchKernelGGL(node_update_gemm, dim3(512), dim3(256), 0, stream,
                     agg, b_in, W_mid, support_m);
  hipMemsetAsync(agg, 0, aggBytes, stream);
  hipLaunchKernelGGL(edge_layer, dim3(8192), dim3(256), 0, stream,
                     edgef, Esrc, Etgt, ee_mid_W1, ee_mid_b1, ee_mid_W2, ee_mid_b2,
                     support_m, agg);
  // ---- layer 3 ----
  hipLaunchKernelGGL(node_out_proj, dim3(256), dim3(256), 0, stream,
                     agg, b_mid, W_out, sup_out);
  hipMemsetAsync(agg3, 0, (size_t)N_NODES * sizeof(float), stream);
  hipLaunchKernelGGL(edge_out_layer, dim3(4096), dim3(256), 0, stream,
                     edgef, Esrc, Etgt, ee_out_W1, ee_out_b1, ee_out_W2, ee_out_b2,
                     sup_out, agg3);
  // ---- pooling ----
  hipMemsetAsync(out, 0, (size_t)out_size * sizeof(float), stream);
  hipLaunchKernelGGL(pool_kernel, dim3(256), dim3(256), 0, stream,
                     agg3, batch, b_out, out);
}
